// RNNAgent_6897717478082
// MI455X (gfx1250) — matrix operand order repaired and checked by reading, not compiled, not run on hardware
//
#include <hip/hip_runtime.h>
#include <math.h>

typedef __bf16 bf16;
typedef __attribute__((ext_vector_type(4)))  bf16  v4bf;   // 8B
typedef __attribute__((ext_vector_type(8)))  bf16  v8bf;   // 16B
typedef __attribute__((ext_vector_type(16))) bf16  v16bf;  // 32B
typedef __attribute__((ext_vector_type(4)))  float v4f;
typedef __attribute__((ext_vector_type(8)))  float v8f;

#define BDIM 4096   // batch
#define IND  1024   // fc1 input width
#define HD   2048   // hidden
#define AD   64     // actions
#define A2D  4096   // A^2

constexpr int BM = 128, BN = 128, BK = 32;
constexpr int LDB = BK + 8;   // 40 bf16 = 80B row stride: 16B-aligned rows, conflict-free banks

__device__ __forceinline__ v8f wmma_bf16(v16bf a, v16bf b, v8f c) {
  return __builtin_amdgcn_wmma_f32_16x16x32_bf16(
      /*neg_a=*/false, a, /*neg_b=*/false, b,
      /*c_mod=*/(short)0, c, /*reuse_a=*/false, /*reuse_b=*/false);
}

// assemble a 16-elem bf16 fragment from two 16B LDS reads
__device__ __forceinline__ v16bf frag16(const bf16* base, int off0, int off1) {
  v16bf r;
  *(v8bf*)&r       = *(const v8bf*)(base + off0);
  *((v8bf*)&r + 1) = *(const v8bf*)(base + off1);
  return r;
}

// async 16B global -> LDS copy (GV mode), tracked by ASYNCcnt
__device__ __forceinline__ void async_ld16(const bf16* lds_dst, const bf16* gsrc) {
  const unsigned loff = (unsigned)(unsigned long long)lds_dst;  // low 32b = LDS offset
  asm volatile("global_load_async_to_lds_b128 %0, %1, off"
               :: "v"(loff), "v"(gsrc)
               : "memory");
}
__device__ __forceinline__ void wait_async0() {
  asm volatile("s_wait_asynccnt 0x0" ::: "memory");
}

// C[M,N] = act( A[M,K] @ W[N,K]^T + bias[N] ), A/W given as bf16 hi/lo planes.
// bf16x3: acc += Ahi*Wlo + Alo*Whi + Ahi*Whi  (fp32 accumulate)
// MODE 0: store f32 C.  MODE 1: relu, then store bf16 hi/lo planes (Chi/Clo).
template <int MODE>
__global__ __launch_bounds__(256)
void gemm_nt_bf16x3(const bf16* __restrict__ Ahi, const bf16* __restrict__ Alo,
                    const bf16* __restrict__ Whi, const bf16* __restrict__ Wlo,
                    const float* __restrict__ bias,
                    float* __restrict__ C,
                    bf16* __restrict__ Chi, bf16* __restrict__ Clo,
                    int M, int N, int K) {
  // double-buffered: [buf][hi/lo][tile]; 2*2*(128*40)*2B * 2 = 80KB static LDS
  __shared__ bf16 sA[2][2][BM * LDB];
  __shared__ bf16 sW[2][2][BN * LDB];

  const int bm   = blockIdx.y * BM;
  const int bn   = blockIdx.x * BN;
  const int tid  = threadIdx.x;
  const int lane = tid & 31;
  const int wave = tid >> 5;     // 0..7
  const int wm   = wave >> 1;    // 0..3 : 32-row group
  const int wn   = wave & 1;     // 0..1 : 64-col group
  const int l16  = lane & 15;
  const int hh   = lane >> 4;    // lane half

  v8f acc[2][4] = {};

  // each plane: 128 rows x 32 bf16 = 512 x 16B chunks / 256 threads = 2 chunks each
  auto issue = [&](int k0, int buf) {
#pragma unroll
    for (int i = 0; i < 2; ++i) {
      const int c = i * 256 + tid;
      const int r = c >> 2;
      const int o = (c & 3) << 3;     // bf16 element offset 0/8/16/24
      const size_t ga = (size_t)(bm + r) * K + k0 + o;
      const size_t gw = (size_t)(bn + r) * K + k0 + o;
      async_ld16(&sA[buf][0][r * LDB + o], &Ahi[ga]);
      async_ld16(&sA[buf][1][r * LDB + o], &Alo[ga]);
      async_ld16(&sW[buf][0][r * LDB + o], &Whi[gw]);
      async_ld16(&sW[buf][1][r * LDB + o], &Wlo[gw]);
    }
  };

  issue(0, 0);
  const int nk = K / BK;
  for (int t = 0; t < nk; ++t) {
    const int cur = t & 1;
    wait_async0();       // my 8 async loads for tile t have landed in LDS
    __syncthreads();     // everyone's landed; everyone done computing tile t-1
    if (t + 1 < nk) issue((t + 1) * BK, cur ^ 1);   // overlaps with compute below

    // A frag (16x32 bf16): lane half h holds K = 8h+0..7 (elems 0-7), 16+8h+0..7 (elems 8-15)
    // B frag (32x16 bf16): lane half h holds K = 8h+0..7 (elems 0-7), 16+8h+0..7 (elems 8-15)
    const int aOff0 = 8 * hh, aOff1 = 16 + 8 * hh;
    const int bOff0 = 8 * hh, bOff1 = 16 + 8 * hh;

    v16bf ah[2], al[2], bh[4], bl[4];
#pragma unroll
    for (int mb = 0; mb < 2; ++mb) {
      const int row = (wm * 32 + mb * 16 + l16) * LDB;
      ah[mb] = frag16(&sA[cur][0][row], aOff0, aOff1);
      al[mb] = frag16(&sA[cur][1][row], aOff0, aOff1);
    }
#pragma unroll
    for (int nb = 0; nb < 4; ++nb) {
      const int row = (wn * 64 + nb * 16 + l16) * LDB;
      bh[nb] = frag16(&sW[cur][0][row], bOff0, bOff1);
      bl[nb] = frag16(&sW[cur][1][row], bOff0, bOff1);
    }
#pragma unroll
    for (int mb = 0; mb < 2; ++mb)
#pragma unroll
      for (int nb = 0; nb < 4; ++nb) {
        acc[mb][nb] = wmma_bf16(ah[mb], bl[nb], acc[mb][nb]);
        acc[mb][nb] = wmma_bf16(al[mb], bh[nb], acc[mb][nb]);
        acc[mb][nb] = wmma_bf16(ah[mb], bh[nb], acc[mb][nb]);
      }
  }

  // epilogue: C layout VGPR i -> row = i + 8*laneHalf, col = lane&15
#pragma unroll
  for (int mb = 0; mb < 2; ++mb) {
#pragma unroll
    for (int nb = 0; nb < 4; ++nb) {
      const int col  = bn + wn * 64 + nb * 16 + l16;
      const float bv = bias[col];
#pragma unroll
      for (int i = 0; i < 8; ++i) {
        const int row = bm + wm * 32 + mb * 16 + hh * 8 + i;
        float vv = acc[mb][nb][i] + bv;
        if (MODE == 0) {
          C[(size_t)row * N + col] = vv;
        } else {
          vv = fmaxf(vv, 0.0f);
          const bf16 h = (bf16)vv;
          Chi[(size_t)row * N + col] = h;
          Clo[(size_t)row * N + col] = (bf16)(vv - (float)h);
        }
      }
    }
  }
}

// fp32 -> (bf16 hi, bf16 lo) split, 4 elements per thread
__global__ __launch_bounds__(256)
void split_bf16(const float* __restrict__ src, bf16* __restrict__ hi,
                bf16* __restrict__ lo, int n4) {
  const int idx = blockIdx.x * 256 + threadIdx.x;
  if (idx >= n4) return;
  const v4f v = ((const v4f*)src)[idx];
  v4bf h, l;
#pragma unroll
  for (int i = 0; i < 4; ++i) {
    h[i] = (bf16)v[i];
    l[i] = (bf16)(v[i] - (float)h[i]);
  }
  ((v4bf*)hi)[idx] = h;
  ((v4bf*)lo)[idx] = l;
}

// GRU gating: h = (1-z)*n + z*h_in ; also emit h as bf16 hi/lo planes for fc2
__global__ __launch_bounds__(256)
void gru_gate(const float* __restrict__ gi, const float* __restrict__ gh,
              const float* __restrict__ h_in, float* __restrict__ h_out,
              bf16* __restrict__ hhi, bf16* __restrict__ hlo) {
  const int idx = blockIdx.x * 256 + threadIdx.x;   // < B*H (exact multiple)
  const int b   = idx >> 11;                         // / H
  const int j   = idx & (HD - 1);
  const size_t base = (size_t)b * 3 * HD + j;
  const float ir = gi[base], iz = gi[base + HD], in_ = gi[base + 2 * HD];
  const float hr = gh[base], hz = gh[base + HD], hn  = gh[base + 2 * HD];
  const float r = 1.0f / (1.0f + __expf(-(ir + hr)));
  const float z = 1.0f / (1.0f + __expf(-(iz + hz)));
  const float n = tanhf(in_ + r * hn);
  const float hv = (1.0f - z) * n + z * h_in[idx];
  h_out[idx] = hv;
  const bf16 hb = (bf16)hv;
  hhi[idx] = hb;
  hlo[idx] = (bf16)(hv - (float)hb);
}

// decode: per batch row, q1[j] = max q[j..j+63]; q2[j] = max q[(j-1+64k) mod 4096]
__global__ __launch_bounds__(64)
void decode_q(const float* __restrict__ q, float* __restrict__ out) {
  __shared__ float s[A2D];
  const int b = blockIdx.x;
  const int t = threadIdx.x;   // 0..63
  const v4f* qr = (const v4f*)(q + (size_t)b * A2D);
#pragma unroll
  for (int i = 0; i < 16; ++i)
    ((v4f*)s)[i * 64 + t] = qr[i * 64 + t];
  __syncthreads();
  float m1 = -3.402823466e+38f, m2 = m1;
#pragma unroll 8
  for (int k = 0; k < 64; ++k) {
    m1 = fmaxf(m1, s[t + k]);
    m2 = fmaxf(m2, s[(t + 64 * k - 1) & (A2D - 1)]);
  }
  out[(size_t)(2 * b) * AD + t]     = m1;
  out[(size_t)(2 * b + 1) * AD + t] = m2;
}

extern "C" void kernel_launch(void* const* d_in, const int* in_sizes, int n_in,
                              void* d_out, int out_size, void* d_ws, size_t ws_size,
                              hipStream_t stream) {
  const float* inputs = (const float*)d_in[0];   // [B, IN]
  const float* hidden = (const float*)d_in[1];   // [B, H]
  const float* W1     = (const float*)d_in[2];   // [H, IN]
  const float* b1     = (const float*)d_in[3];   // [H]
  const float* Wih    = (const float*)d_in[4];   // [3H, H]
  const float* Whh    = (const float*)d_in[5];   // [3H, H]
  const float* bih    = (const float*)d_in[6];   // [3H]
  const float* bhh    = (const float*)d_in[7];   // [3H]
  const float* W2     = (const float*)d_in[8];   // [A2, H]
  const float* b2     = (const float*)d_in[9];   // [A2]

  float* out   = (float*)d_out;                  // [2B, A]
  float* h_out = out + (size_t)2 * BDIM * AD;    // [B, H] (2nd output)

  // ---- workspace layout ----
  char* p = (char*)d_ws;
  auto take = [&](size_t bytes) { char* r = p; p += (bytes + 255) & ~(size_t)255; return r; };
  float* gi   = (float*)take((size_t)BDIM * 3 * HD * 4);   // [B,3H]
  float* gh   = (float*)take((size_t)BDIM * 3 * HD * 4);   // [B,3H]
  float* q    = gi;                                        // reuse after gating: [B,A2]
  bf16* ihi   = (bf16*)take((size_t)BDIM * IND * 2);
  bf16* ilo   = (bf16*)take((size_t)BDIM * IND * 2);
  bf16* xhi   = (bf16*)take((size_t)BDIM * HD * 2);
  bf16* xlo   = (bf16*)take((size_t)BDIM * HD * 2);
  bf16* shi   = (bf16*)take((size_t)BDIM * HD * 2);        // hidden_state split
  bf16* slo   = (bf16*)take((size_t)BDIM * HD * 2);
  bf16* hhi   = (bf16*)take((size_t)BDIM * HD * 2);        // new h split
  bf16* hlo   = (bf16*)take((size_t)BDIM * HD * 2);
  bf16* w1hi  = (bf16*)take((size_t)HD * IND * 2);
  bf16* w1lo  = (bf16*)take((size_t)HD * IND * 2);
  bf16* wihhi = (bf16*)take((size_t)3 * HD * HD * 2);
  bf16* wihlo = (bf16*)take((size_t)3 * HD * HD * 2);
  bf16* whhhi = (bf16*)take((size_t)3 * HD * HD * 2);
  bf16* whhlo = (bf16*)take((size_t)3 * HD * HD * 2);
  bf16* w2hi  = (bf16*)take((size_t)A2D * HD * 2);
  bf16* w2lo  = (bf16*)take((size_t)A2D * HD * 2);

  const dim3 blk(256);
  auto g1 = [](int n4) { return dim3((unsigned)((n4 + 255) / 256)); };

  // ---- fp32 -> bf16 hi/lo pre-splits (memory bound) ----
  split_bf16<<<g1(BDIM * IND / 4), blk, 0, stream>>>(inputs, ihi, ilo, BDIM * IND / 4);
  split_bf16<<<g1(BDIM * HD / 4),  blk, 0, stream>>>(hidden, shi, slo, BDIM * HD / 4);
  split_bf16<<<g1(HD * IND / 4),   blk, 0, stream>>>(W1,  w1hi,  w1lo,  HD * IND / 4);
  split_bf16<<<g1(3 * HD * HD / 4), blk, 0, stream>>>(Wih, wihhi, wihlo, 3 * HD * HD / 4);
  split_bf16<<<g1(3 * HD * HD / 4), blk, 0, stream>>>(Whh, whhhi, whhlo, 3 * HD * HD / 4);
  split_bf16<<<g1(A2D * HD / 4),   blk, 0, stream>>>(W2,  w2hi,  w2lo,  A2D * HD / 4);

  // ---- x = relu(inputs @ W1^T + b1), emitted directly as bf16 hi/lo ----
  gemm_nt_bf16x3<1><<<dim3(HD / BN, BDIM / BM), blk, 0, stream>>>(
      ihi, ilo, w1hi, w1lo, b1, nullptr, xhi, xlo, BDIM, HD, IND);
  // ---- gi = x @ Wih^T + bih ; gh = h_in @ Whh^T + bhh (f32) ----
  gemm_nt_bf16x3<0><<<dim3(3 * HD / BN, BDIM / BM), blk, 0, stream>>>(
      xhi, xlo, wihhi, wihlo, bih, gi, nullptr, nullptr, BDIM, 3 * HD, HD);
  gemm_nt_bf16x3<0><<<dim3(3 * HD / BN, BDIM / BM), blk, 0, stream>>>(
      shi, slo, whhhi, whhlo, bhh, gh, nullptr, nullptr, BDIM, 3 * HD, HD);
  // ---- gating -> h (f32 into d_out slot) + bf16 hi/lo for fc2 ----
  gru_gate<<<dim3(BDIM * HD / 256), blk, 0, stream>>>(gi, gh, hidden, h_out, hhi, hlo);
  // ---- q = h @ W2^T + b2 (f32, reuses gi region) ----
  gemm_nt_bf16x3<0><<<dim3(A2D / BN, BDIM / BM), blk, 0, stream>>>(
      hhi, hlo, w2hi, w2lo, b2, q, nullptr, nullptr, BDIM, A2D, HD);
  // ---- windowed maxes -> interleaved [2B, A] ----
  decode_q<<<dim3(BDIM), dim3(64), 0, stream>>>(q, out);

  (void)in_sizes; (void)n_in; (void)out_size; (void)ws_size;
}
